// AttentionPerformer_72146860638368
// MI455X (gfx1250) — matrix operand order repaired and checked by reading, not compiled, not run on hardware
//
#include <hip/hip_runtime.h>
#include <hip/hip_bf16.h>

typedef __bf16 bf16_t;
typedef __attribute__((ext_vector_type(8)))  __bf16 v8bf;
typedef __attribute__((ext_vector_type(16))) __bf16 v16bf;
typedef __attribute__((ext_vector_type(8)))  float  v8f;
typedef int v4i __attribute__((vector_size(16)));

constexpr int B_ = 4, N_ = 3136, C_ = 768, H_ = 12, DH_ = 64, M_ = 384;
constexpr int R_ = B_ * N_;       // 12544 rows (multiple of 32)
constexpr int O_ = 3 * C_;        // 2304 qkv outputs
constexpr float EPS_   = 1e-8f;
constexpr float RATIO_ = 0.22589944f;  // 384^-0.25
constexpr float INV16_ = 0.0625f;      // 0.5 * (64^-0.25)^2
constexpr float INV8_  = 0.125f;       // 1/sqrt(num_realizations=64)

// ---- CDNA5 async load-to-LDS (guarded; falls back to reg-staged copy) ----
#if defined(__AMDGCN__) && __has_builtin(__builtin_amdgcn_global_load_async_to_lds_b128) && \
    __has_builtin(__builtin_amdgcn_s_wait_asynccnt)
#define HAVE_ASYNC 1
typedef __attribute__((address_space(1))) v4i g_v4i;
typedef __attribute__((address_space(3))) v4i l_v4i;
#else
#define HAVE_ASYNC 0
#endif

__device__ __forceinline__ void stage16(const bf16_t* g, bf16_t* l) {
#if HAVE_ASYNC
  __builtin_amdgcn_global_load_async_to_lds_b128((g_v4i*)g, (l_v4i*)l, 0, 0);
#else
  *(v8bf*)l = *(const v8bf*)g;
#endif
}
__device__ __forceinline__ void wait_async_1() {
#if HAVE_ASYNC
  __builtin_amdgcn_s_wait_asynccnt(1);
#endif
}
__device__ __forceinline__ void wait_async_0() {
#if HAVE_ASYNC
  __builtin_amdgcn_s_wait_asynccnt(0);
#endif
}

// ---- WMMA fragment helpers (wave32, 16x16x32 bf16) ----
// A 16xK tile: lane l holds row l%16; k-chunk (l>>4)*8 .. +7 and +16 .. +23
__device__ __forceinline__ v16bf ldfragA(const bf16_t* p) {
  union { v16bf v; struct { v8bf lo, hi; } s; } u;
  u.s.lo = *(const v8bf*)(p);
  u.s.hi = *(const v8bf*)(p + 16);
  return u.v;
}
// B Kx16 tile: lane l holds column l%16; k-chunk (l>>4)*8 .. +7 and +16 .. +23
__device__ __forceinline__ v16bf ldfragB(const bf16_t* p) {
  union { v16bf v; struct { v8bf lo, hi; } s; } u;
  u.s.lo = *(const v8bf*)(p);
  u.s.hi = *(const v8bf*)(p + 16);
  return u.v;
}
__device__ __forceinline__ v8f wmma_bf16(v16bf a, v16bf b, v8f c) {
  return __builtin_amdgcn_wmma_f32_16x16x32_bf16(false, a, false, b, (short)0, c,
                                                 false, false);
}
// order-preserving float <-> uint key for atomicMax-based global max
__device__ __forceinline__ unsigned fkey(float f) {
  unsigned u = __float_as_uint(f);
  return u ^ ((unsigned)((int)u >> 31) | 0x80000000u);
}
__device__ __forceinline__ float funkey(unsigned k) {
  unsigned bits = (k & 0x80000000u) ? (k ^ 0x80000000u) : ~k;
  return __uint_as_float(bits);
}

// ---------------- kernel 0: weight conversion ----------------
__global__ void k_convert_weights(const float* __restrict__ qkvw,
                                  const float* __restrict__ projw,
                                  const float* __restrict__ w,
                                  bf16_t* __restrict__ qkvw_b,
                                  bf16_t* __restrict__ projw_b,
                                  bf16_t* __restrict__ w_b) {
  int i = blockIdx.x * blockDim.x + threadIdx.x;
  if (i < O_ * C_)  qkvw_b[i]  = (bf16_t)qkvw[i];
  if (i < C_ * C_)  projw_b[i] = (bf16_t)projw[i];
  if (i < M_ * DH_) w_b[i]     = (bf16_t)w[i];
}

// ---------------- kernel 1: LayerNorm -> bf16 ----------------
__global__ void k_layernorm(const float* __restrict__ x, const float* __restrict__ g,
                            const float* __restrict__ b, bf16_t* __restrict__ h) {
  int row = blockIdx.x;
  const float* xr = x + (size_t)row * C_;
  float s = 0.f, s2 = 0.f;
  for (int i = threadIdx.x; i < C_; i += blockDim.x) { float v = xr[i]; s += v; s2 += v * v; }
  for (int off = 16; off > 0; off >>= 1) { s += __shfl_xor(s, off); s2 += __shfl_xor(s2, off); }
  __shared__ float red[16];
  __shared__ float stats[2];
  int wid = threadIdx.x >> 5;
  if ((threadIdx.x & 31) == 0) { red[wid] = s; red[8 + wid] = s2; }
  __syncthreads();
  if (threadIdx.x == 0) {
    float ts = 0.f, ts2 = 0.f;
    for (int i = 0; i < (int)(blockDim.x >> 5); ++i) { ts += red[i]; ts2 += red[8 + i]; }
    float mu = ts / C_;
    float var = ts2 / C_ - mu * mu;
    stats[0] = mu; stats[1] = rsqrtf(var + 1e-5f);
  }
  __syncthreads();
  float mu = stats[0], rs = stats[1];
  for (int i = threadIdx.x; i < C_; i += blockDim.x)
    h[(size_t)row * C_ + i] = (bf16_t)((xr[i] - mu) * rs * g[i] + b[i]);
}

// ---------------- kernel 2: QKV GEMM (WMMA, 32x64/wave, async LDS A-panel) ----------------
__global__ void __launch_bounds__(128, 1)
k_qkv_gemm(const bf16_t* __restrict__ hb, const bf16_t* __restrict__ qw,
           const float* __restrict__ bias,
           bf16_t* __restrict__ qo, bf16_t* __restrict__ ko,
           float* __restrict__ vfo, bf16_t* __restrict__ vto) {
  __shared__ alignas(16) bf16_t apanel[2][32 * 40];   // 32 rows x 32 k, stride 40 (pad)
  int lane = threadIdx.x & 31, wv = threadIdx.x >> 5;
  int r0 = blockIdx.x * 32;
  int c0 = (blockIdx.y * 4 + wv) * 64;
  int aoff = (lane >> 4) << 3;
  int boff = (lane >> 4) << 3;
  int srow = threadIdx.x >> 2, schunk = (threadIdx.x & 3) << 3;  // 128 thr cover 32x32 panel
  const bf16_t* gstage = hb + (size_t)(r0 + srow) * C_ + schunk;
  int lidx = srow * 40 + schunk;
  const bf16_t* bp = qw + (size_t)(c0 + (lane & 15)) * C_ + boff;
  v8f acc[2][4] = {};
  stage16(gstage, &apanel[0][0] + lidx);
  for (int kk = 0; kk < C_ / 32; ++kk) {
    int k = kk * 32;
    int cur = kk & 1;
    if (kk + 1 < C_ / 32) {
      stage16(gstage + k + 32, &apanel[cur ^ 1][0] + lidx);
      wait_async_1();
    } else {
      wait_async_0();
    }
    __syncthreads();
    const bf16_t* a0p = &apanel[cur][0] + (size_t)(lane & 15) * 40 + aoff;
    v16bf a0 = ldfragA(a0p);
    v16bf a1 = ldfragA(a0p + 16 * 40);
    for (int j = 0; j < 4; ++j) {
      v16bf b = ldfragB(bp + k + (size_t)j * 16 * C_);
      acc[0][j] = wmma_bf16(a0, b, acc[0][j]);
      acc[1][j] = wmma_bf16(a1, b, acc[1][j]);
    }
    __syncthreads();
  }
  int half = lane >> 4;
  for (int j = 0; j < 4; ++j) {
    int col = c0 + j * 16 + (lane & 15);
    int s = col / C_;
    int within = col - s * C_;
    int hh = within >> 6, dd = within & 63;
    float bv = bias[col];
    for (int i = 0; i < 2; ++i) {
      for (int g = 0; g < 8; ++g) {
        int row = r0 + i * 16 + half * 8 + g;
        int bb = row / N_, nn = row - bb * N_;
        float val = acc[i][j][g] + bv;
        size_t bh = (size_t)bb * H_ + hh;
        if (s == 0)      qo[(bh * N_ + nn) * DH_ + dd] = (bf16_t)val;
        else if (s == 1) ko[(bh * N_ + nn) * DH_ + dd] = (bf16_t)val;
        else {
          vfo[(size_t)row * C_ + within] = val;
          vto[(bh * DH_ + dd) * N_ + nn] = (bf16_t)val;
        }
      }
    }
  }
}

// ---------------- kernel 3: diag terms ----------------
__global__ void k_diag(const bf16_t* __restrict__ qb, const bf16_t* __restrict__ kb,
                       float* __restrict__ dq, float* __restrict__ dk) {
  int r = blockIdx.x * blockDim.x + threadIdx.x;
  if (r >= B_ * H_ * N_) return;
  const bf16_t* qp = qb + (size_t)r * DH_;
  const bf16_t* kp = kb + (size_t)r * DH_;
  float sq = 0.f, sk = 0.f;
  for (int d = 0; d < DH_; ++d) {
    float a = (float)qp[d]; sq += a * a;
    float c = (float)kp[d]; sk += c * c;
  }
  dq[r] = sq * INV16_;
  dk[r] = sk * INV16_;
}

// ---------------- kernel 4a: global max of dash_k (WMMA) ----------------
__global__ void __launch_bounds__(128, 1)
k_kmax(const bf16_t* __restrict__ kin, const bf16_t* __restrict__ wfeat,
       unsigned* __restrict__ gmax) {
  int lane = threadIdx.x & 31, wv = threadIdx.x >> 5;
  int nt = blockIdx.x, bh = blockIdx.y;
  int aoff = (lane >> 4) << 3;
  int boff = (lane >> 4) << 3;
  const bf16_t* ap = kin + ((size_t)bh * N_ + nt * 16 + (lane & 15)) * DH_ + aoff;
  v16bf a0 = ldfragA(ap), a1 = ldfragA(ap + 32);
  float mx = -3.0e38f;
  for (int mt = wv; mt < M_ / 16; mt += 4) {
    const bf16_t* bp = wfeat + (size_t)(mt * 16 + (lane & 15)) * DH_ + boff;
    v8f c = {};
    c = wmma_bf16(a0, ldfragB(bp), c);
    c = wmma_bf16(a1, ldfragB(bp + 32), c);
    for (int g = 0; g < 8; ++g) mx = fmaxf(mx, c[g]);
  }
  for (int off = 16; off > 0; off >>= 1) mx = fmaxf(mx, __shfl_xor(mx, off));
  if (lane == 0) atomicMax(gmax, fkey(mx));
}

// ---------------- kernel 4b: kp = ratio*(exp(dash-diag-mx)+eps), transposed ----------------
__global__ void __launch_bounds__(128, 1)
k_kp(const bf16_t* __restrict__ kin, const bf16_t* __restrict__ wfeat,
     const float* __restrict__ dk, const unsigned* __restrict__ gmax,
     bf16_t* __restrict__ kpT) {
  int lane = threadIdx.x & 31, wv = threadIdx.x >> 5;
  int nt = blockIdx.x, bh = blockIdx.y;
  int half = lane >> 4;
  int aoff = half << 3;
  int boff = half << 3;
  const bf16_t* ap = kin + ((size_t)bh * N_ + nt * 16 + (lane & 15)) * DH_ + aoff;
  v16bf a0 = ldfragA(ap), a1 = ldfragA(ap + 32);
  float mx = funkey(*gmax);
  int n_base = nt * 16 + half * 8;
  float dg[8];
  for (int g = 0; g < 8; ++g) dg[g] = dk[(size_t)bh * N_ + n_base + g];
  for (int mt = wv; mt < M_ / 16; mt += 4) {
    const bf16_t* bp = wfeat + (size_t)(mt * 16 + (lane & 15)) * DH_ + boff;
    v8f c = {};
    c = wmma_bf16(a0, ldfragB(bp), c);
    c = wmma_bf16(a1, ldfragB(bp + 32), c);
    v8bf ov;
    for (int g = 0; g < 8; ++g)
      ov[g] = (bf16_t)(RATIO_ * (__expf(c[g] - dg[g] - mx) + EPS_));
    int m = mt * 16 + (lane & 15);
    *(v8bf*)(kpT + ((size_t)bh * M_ + m) * N_ + n_base) = ov;   // contiguous store
  }
}

// ---------------- kernel 5a: kp_sum over n ----------------
__global__ void k_kpsum(const bf16_t* __restrict__ kpT, float* __restrict__ ksum) {
  int lane = threadIdx.x & 31, wv = threadIdx.x >> 5;
  int row = blockIdx.x * (blockDim.x >> 5) + wv;
  if (row >= B_ * H_ * M_) return;
  const bf16_t* p = kpT + (size_t)row * N_;
  float s = 0.f;
  for (int i = lane; i < N_; i += 32) s += (float)p[i];
  for (int off = 16; off > 0; off >>= 1) s += __shfl_xor(s, off);
  if (lane == 0) ksum[row] = s;
}

// ---------------- kernel 5b: kptv[d,m] = sum_n vT[d,n]*kpT[m,n] (WMMA 32x64/wave) ----------------
__global__ void __launch_bounds__(128, 1)
k_kptv(const bf16_t* __restrict__ vT, const bf16_t* __restrict__ kpT,
       bf16_t* __restrict__ kptv) {
  int lane = threadIdx.x & 31, wv = threadIdx.x >> 5;
  int t = blockIdx.x * 4 + wv;            // 12 tiles of 32x64 per head
  int bh = blockIdx.y;
  int rt = (t & 1) * 32;                  // d block
  int ct = (t >> 1) * 64;                 // m block
  int aoff = (lane >> 4) << 3;
  int boff = (lane >> 4) << 3;
  const bf16_t* ap = vT  + ((size_t)bh * DH_ + rt + (lane & 15)) * N_ + aoff;
  const bf16_t* bp = kpT + ((size_t)bh * M_  + ct + (lane & 15)) * N_ + boff;
  v8f acc[2][4] = {};
  for (int k = 0; k < N_; k += 32) {
    if ((k & 255) == 0) {
      __builtin_prefetch(ap + k + 512, 0, 1);
      __builtin_prefetch(bp + k + 512, 0, 1);
    }
    v16bf a0 = ldfragA(ap + k);
    v16bf a1 = ldfragA(ap + (size_t)16 * N_ + k);
    for (int j = 0; j < 4; ++j) {
      v16bf b = ldfragB(bp + (size_t)j * 16 * N_ + k);
      acc[0][j] = wmma_bf16(a0, b, acc[0][j]);
      acc[1][j] = wmma_bf16(a1, b, acc[1][j]);
    }
  }
  int half = lane >> 4;
  for (int i = 0; i < 2; ++i)
    for (int j = 0; j < 4; ++j) {
      int m = ct + j * 16 + (lane & 15);
      for (int g = 0; g < 8; ++g) {
        int d = rt + i * 16 + half * 8 + g;
        kptv[((size_t)bh * DH_ + d) * M_ + m] = (bf16_t)acc[i][j][g];
      }
    }
}

// ---------------- kernel 6: fused qp / Dn / y (WMMA, LDS re-stage) ----------------
__global__ void __launch_bounds__(128, 1)
k_qy(const bf16_t* __restrict__ qb, const bf16_t* __restrict__ wfeat,
     const float* __restrict__ dq, const float* __restrict__ ksum,
     const bf16_t* __restrict__ kptv, bf16_t* __restrict__ ycat) {
  __shared__ alignas(16) bf16_t stage[4][16 * 32];
  int lane = threadIdx.x & 31, wv = threadIdx.x >> 5;
  int nt = blockIdx.x * 4 + wv;
  int bh = blockIdx.y;
  int bb = bh / H_, hh = bh - bb * H_;
  int half = lane >> 4;
  int aoff = half << 3;
  int boff = half << 3;
  int n0 = nt * 16;
  const bf16_t* ap = qb + ((size_t)bh * N_ + n0 + (lane & 15)) * DH_ + aoff;
  v16bf a0 = ldfragA(ap), a1 = ldfragA(ap + 32);

  // pass 1: per-row max over all M
  float mx8[8];
  for (int g = 0; g < 8; ++g) mx8[g] = -3.0e38f;
  for (int mt = 0; mt < M_ / 16; ++mt) {
    const bf16_t* bp = wfeat + (size_t)(mt * 16 + (lane & 15)) * DH_ + boff;
    v8f c = {};
    c = wmma_bf16(a0, ldfragB(bp), c);
    c = wmma_bf16(a1, ldfragB(bp + 32), c);
    for (int g = 0; g < 8; ++g) mx8[g] = fmaxf(mx8[g], c[g]);
  }
  for (int off = 1; off < 16; off <<= 1)
    for (int g = 0; g < 8; ++g) mx8[g] = fmaxf(mx8[g], __shfl_xor(mx8[g], off));
  float dg8[8];
  for (int g = 0; g < 8; ++g) dg8[g] = dq[(size_t)bh * N_ + n0 + half * 8 + g];

  // pass 2: recompute dash, exp -> qp, accumulate Dn and y = qp @ kptv^T
  bf16_t* st = stage[wv];
  const bf16_t* kvb = kptv + (size_t)bh * DH_ * M_;
  const float* ks = ksum + (size_t)bh * M_;
  v8f y[4] = {};
  float dn8[8] = {};
  for (int mp = 0; mp < M_ / 32; ++mp) {
    for (int sub = 0; sub < 2; ++sub) {
      int mt = mp * 2 + sub;
      const bf16_t* bp = wfeat + (size_t)(mt * 16 + (lane & 15)) * DH_ + boff;
      v8f c = {};
      c = wmma_bf16(a0, ldfragB(bp), c);
      c = wmma_bf16(a1, ldfragB(bp + 32), c);
      float kscol = ks[mt * 16 + (lane & 15)];
      for (int g = 0; g < 8; ++g) {
        float e = RATIO_ * (__expf(c[g] - dg8[g] - mx8[g]) + EPS_);
        dn8[g] += e * kscol;
        st[(half * 8 + g) * 32 + sub * 16 + (lane & 15)] = (bf16_t)e;
      }
    }
    v16bf aq = ldfragA(st + (size_t)(lane & 15) * 32 + aoff);   // C-layout -> A-layout via LDS
    int m0 = mp * 32;
    for (int dt = 0; dt < 4; ++dt) {
      const bf16_t* bp = kvb + (size_t)(dt * 16 + (lane & 15)) * M_ + m0 + boff;
      y[dt] = wmma_bf16(aq, ldfragB(bp), y[dt]);
    }
  }
  for (int off = 1; off < 16; off <<= 1)
    for (int g = 0; g < 8; ++g) dn8[g] += __shfl_xor(dn8[g], off);
  for (int dt = 0; dt < 4; ++dt) {
    int d = dt * 16 + (lane & 15);
    for (int g = 0; g < 8; ++g) {
      int n = n0 + half * 8 + g;
      float val = y[dt][g] / (dn8[g] + EPS_) * INV8_;
      ycat[((size_t)bb * N_ + n) * C_ + hh * DH_ + d] = (bf16_t)val;
    }
  }
}

// ---------------- kernel 7: output projection + residual (WMMA 32x64/wave, async LDS) ----------------
__global__ void __launch_bounds__(128, 1)
k_proj(const bf16_t* __restrict__ yb, const bf16_t* __restrict__ pw,
       const float* __restrict__ pb, const float* __restrict__ vf,
       float* __restrict__ out) {
  __shared__ alignas(16) bf16_t apanel[2][32 * 40];
  int lane = threadIdx.x & 31, wv = threadIdx.x >> 5;
  int r0 = blockIdx.x * 32;
  int c0 = (blockIdx.y * 4 + wv) * 64;
  int aoff = (lane >> 4) << 3;
  int boff = (lane >> 4) << 3;
  int srow = threadIdx.x >> 2, schunk = (threadIdx.x & 3) << 3;
  const bf16_t* gstage = yb + (size_t)(r0 + srow) * C_ + schunk;
  int lidx = srow * 40 + schunk;
  const bf16_t* bp = pw + (size_t)(c0 + (lane & 15)) * C_ + boff;
  v8f acc[2][4] = {};
  stage16(gstage, &apanel[0][0] + lidx);
  for (int kk = 0; kk < C_ / 32; ++kk) {
    int k = kk * 32;
    int cur = kk & 1;
    if (kk + 1 < C_ / 32) {
      stage16(gstage + k + 32, &apanel[cur ^ 1][0] + lidx);
      wait_async_1();
    } else {
      wait_async_0();
    }
    __syncthreads();
    const bf16_t* a0p = &apanel[cur][0] + (size_t)(lane & 15) * 40 + aoff;
    v16bf a0 = ldfragA(a0p);
    v16bf a1 = ldfragA(a0p + 16 * 40);
    for (int j = 0; j < 4; ++j) {
      v16bf b = ldfragB(bp + k + (size_t)j * 16 * C_);
      acc[0][j] = wmma_bf16(a0, b, acc[0][j]);
      acc[1][j] = wmma_bf16(a1, b, acc[1][j]);
    }
    __syncthreads();
  }
  int half = lane >> 4;
  for (int j = 0; j < 4; ++j) {
    int col = c0 + j * 16 + (lane & 15);
    float bias = pb[col];
    for (int i = 0; i < 2; ++i) {
      for (int g = 0; g < 8; ++g) {
        size_t idx = (size_t)(r0 + i * 16 + half * 8 + g) * C_ + col;
        out[idx] = vf[idx] + acc[i][j][g] + bias;
      }
    }
  }
}

extern "C" void kernel_launch(void* const* d_in, const int* in_sizes, int n_in,
                              void* d_out, int out_size, void* d_ws, size_t ws_size,
                              hipStream_t stream) {
  const float* x      = (const float*)d_in[0];
  const float* ln_w   = (const float*)d_in[1];
  const float* ln_b   = (const float*)d_in[2];
  const float* qkv_w  = (const float*)d_in[3];
  const float* qkv_b  = (const float*)d_in[4];
  const float* proj_w = (const float*)d_in[5];
  const float* proj_b = (const float*)d_in[6];
  const float* w      = (const float*)d_in[7];
  float* out = (float*)d_out;
  (void)in_sizes; (void)n_in; (void)out_size; (void)ws_size;

  char* ws = (char*)d_ws;
  size_t off = 0;
  auto alloc = [&](size_t bytes) {
    char* p = ws + off;
    off = (off + bytes + 255) & ~(size_t)255;
    return p;
  };
  bf16_t* hb      = (bf16_t*)alloc((size_t)R_ * C_ * 2);
  bf16_t* qkvw_b  = (bf16_t*)alloc((size_t)O_ * C_ * 2);
  bf16_t* projw_b = (bf16_t*)alloc((size_t)C_ * C_ * 2);
  bf16_t* w_b     = (bf16_t*)alloc((size_t)M_ * DH_ * 2);
  bf16_t* qb      = (bf16_t*)alloc((size_t)B_ * H_ * N_ * DH_ * 2);
  bf16_t* kb      = (bf16_t*)alloc((size_t)B_ * H_ * N_ * DH_ * 2);
  bf16_t* vT      = (bf16_t*)alloc((size_t)B_ * H_ * N_ * DH_ * 2);
  float*  vf      = (float*) alloc((size_t)R_ * C_ * 4);
  float*  dq      = (float*) alloc((size_t)B_ * H_ * N_ * 4);
  float*  dk      = (float*) alloc((size_t)B_ * H_ * N_ * 4);
  unsigned* gmax  = (unsigned*)alloc(256);
  bf16_t* kpT     = (bf16_t*)alloc((size_t)B_ * H_ * M_ * N_ * 2);
  float*  ksum    = (float*) alloc((size_t)B_ * H_ * M_ * 4);
  bf16_t* kptv    = (bf16_t*)alloc((size_t)B_ * H_ * DH_ * M_ * 2);
  bf16_t* ycat    = (bf16_t*)alloc((size_t)R_ * C_ * 2);

  (void)hipMemsetAsync(gmax, 0, 4, stream);   // key 0 == -inf under fkey ordering
  k_convert_weights<<<(O_ * C_ + 255) / 256, 256, 0, stream>>>(
      qkv_w, proj_w, w, qkvw_b, projw_b, w_b);
  k_layernorm<<<R_, 256, 0, stream>>>(x, ln_w, ln_b, hb);
  k_qkv_gemm<<<dim3(R_ / 32, O_ / 256), 128, 0, stream>>>(
      hb, qkvw_b, qkv_b, qb, kb, vf, vT);
  k_diag<<<(B_ * H_ * N_ + 255) / 256, 256, 0, stream>>>(qb, kb, dq, dk);
  k_kmax<<<dim3(N_ / 16, B_ * H_), 128, 0, stream>>>(kb, w_b, gmax);
  k_kp<<<dim3(N_ / 16, B_ * H_), 128, 0, stream>>>(kb, w_b, dk, gmax, kpT);
  k_kpsum<<<(B_ * H_ * M_ + 7) / 8, 256, 0, stream>>>(kpT, ksum);
  k_kptv<<<dim3(3, B_ * H_), 128, 0, stream>>>(vT, kpT, kptv);
  k_qy<<<dim3(N_ / 16 / 4, B_ * H_), 128, 0, stream>>>(qb, w_b, dq, ksum, kptv, ycat);
  k_proj<<<dim3(R_ / 32, C_ / 256), 128, 0, stream>>>(ycat, projw_b, proj_b, vf, out);
}
